// MultiHeadAttention_16844861735789
// MI455X (gfx1250) — matrix operand order repaired and checked by reading, not compiled, not run on hardware
//
#include <hip/hip_runtime.h>

// ---------------------------------------------------------------------------
// MultiHeadAttention forward for MI455X (gfx1250, wave32, WMMA bf16 path)
// B=4, S=2048, D=1024, H=16, DH=64
// ---------------------------------------------------------------------------

typedef __attribute__((ext_vector_type(16))) __bf16 v16bf;
typedef __attribute__((ext_vector_type(8)))  float  v8f;

struct U128 { unsigned int v[4]; };          // 16 bytes
struct Frag32 { U128 lo, hi; };              // 32 bytes -> v16bf
static_assert(sizeof(Frag32) == 32, "frag size");

__device__ __forceinline__ v16bf frag_cast(const Frag32& f) {
    return __builtin_bit_cast(v16bf, f);
}

__device__ __forceinline__ unsigned short f2bf(float f) {
    unsigned int u = __builtin_bit_cast(unsigned int, f);
    u += 0x7FFFu + ((u >> 16) & 1u);         // round-to-nearest-even
    return (unsigned short)(u >> 16);
}

__device__ __forceinline__ v8f v8f_zero() {
    v8f z = {0.f, 0.f, 0.f, 0.f, 0.f, 0.f, 0.f, 0.f};
    return z;
}

// same-wave LDS store->load ordering (per-wave scratch; no block barrier needed)
__device__ __forceinline__ void wait_lds() {
    asm volatile("s_wait_dscnt 0x0" ::: "memory");
}

constexpr int Bb = 4, S = 2048, D = 1024, H = 16, DH = 64;
constexpr int M_TOT = Bb * S;                // 8192

// ---------------------------------------------------------------------------
// f32 -> bf16 conversion (4 elements / thread)
// ---------------------------------------------------------------------------
__global__ __launch_bounds__(256)
void f32_to_bf16_kernel(const float* __restrict__ in,
                        unsigned short* __restrict__ out, int n4) {
    int i = (blockIdx.x * 256 + threadIdx.x) * 4;
    if (i < n4) {
        float4 f = *reinterpret_cast<const float4*>(in + i);
        unsigned short r[4] = { f2bf(f.x), f2bf(f.y), f2bf(f.z), f2bf(f.w) };
        *reinterpret_cast<uint2*>(out + i) = *reinterpret_cast<uint2*>(r);
    }
}

// ---------------------------------------------------------------------------
// Tiled bf16 GEMM:  Y[m,n] = sum_k A[m,k] * W[n,k] + bias[n]
// Block tile 128(M) x 256(N), 8 waves as 2M x 4N, wave tile 64x64
// (acc[4][4], 16 WMMA per 32-deep slice from 8 fragment loads).
// Double-buffered LDS: ONE barrier per K stage; stage i+1 stores go to the
// other buffer, and the top-of-stage barrier guarantees buffer p's readers
// finished before stage i+2 overwrites it. Global loads pipelined one stage
// ahead into registers.
// MODE 0: bf16 row-major out [M,N]
// MODE 1: bf16 transposed V out [B,H,DH,S]
// MODE 2: f32 row-major out [M,N]
// ---------------------------------------------------------------------------
template <int MODE>
__global__ __launch_bounds__(256)
void gemm_bf16_kernel(const unsigned short* __restrict__ A,
                      const unsigned short* __restrict__ W,
                      const float* __restrict__ bias,
                      void* __restrict__ outp,
                      int M, int N, int K) {
    constexpr int KT  = 32;                          // k per stage
    constexpr int LDT = 40;                          // padded k-stride (bf16)
    __shared__ __align__(16) unsigned short ldsA[2][128 * LDT];   // 20 KB
    __shared__ __align__(16) unsigned short ldsB[2][256 * LDT];   // 40 KB

    const int tid   = threadIdx.x;
    const int lane  = tid & 31;
    const int wv    = tid >> 5;
    const int hi    = lane >> 4;
    const int ln16  = lane & 15;
    const int waveM = wv & 1;                        // 0..1 (64 rows each)
    const int waveN = wv >> 1;                       // 0..3 (64 cols each)
    const int m0 = blockIdx.y * 128;
    const int n0 = blockIdx.x * 256;

    v8f acc[4][4];
#pragma unroll
    for (int mt = 0; mt < 4; ++mt)
#pragma unroll
        for (int nt = 0; nt < 4; ++nt) acc[mt][nt] = v8f_zero();

    // staging: A tile 128x32 -> 2 threads/row; B tile 256x32 -> 1 thread/row
    const int arow = tid >> 1;                       // 0..127
    const int alk  = (tid & 1) * 16;                 // 0 or 16 (elements)
    const int brow = tid;                            // 0..255

    const unsigned short* gA = A + (size_t)(m0 + arow) * K + alk;
    const unsigned short* gB = W + (size_t)(n0 + brow) * K;

    U128 ra[2], rb[4];
#pragma unroll
    for (int j = 0; j < 2; ++j) ra[j] = reinterpret_cast<const U128*>(gA)[j];
#pragma unroll
    for (int j = 0; j < 4; ++j) rb[j] = reinterpret_cast<const U128*>(gB)[j];

    int p = 0;
    for (int k0 = 0; k0 < K; k0 += KT, p ^= 1) {
        // stage current tiles into LDS buffer p
#pragma unroll
        for (int j = 0; j < 2; ++j)
            *reinterpret_cast<U128*>(&ldsA[p][arow * LDT + alk + j * 8]) = ra[j];
#pragma unroll
        for (int j = 0; j < 4; ++j)
            *reinterpret_cast<U128*>(&ldsB[p][brow * LDT + j * 8]) = rb[j];
        __syncthreads();

        // issue next stage's global loads early (overlap with WMMA below)
        if (k0 + KT < K) {
            const unsigned short* nA = gA + (k0 + KT);
            const unsigned short* nB = gB + (k0 + KT);
#pragma unroll
            for (int j = 0; j < 2; ++j)
                ra[j] = reinterpret_cast<const U128*>(nA)[j];
#pragma unroll
            for (int j = 0; j < 4; ++j)
                rb[j] = reinterpret_cast<const U128*>(nB)[j];
        }

        v16bf afrag[4];
#pragma unroll
        for (int mt = 0; mt < 4; ++mt) {
            int r = waveM * 64 + mt * 16 + ln16;
            Frag32 f;
            f.lo = *reinterpret_cast<const U128*>(&ldsA[p][r * LDT + hi * 8]);
            f.hi = *reinterpret_cast<const U128*>(&ldsA[p][r * LDT + 16 + hi * 8]);
            afrag[mt] = frag_cast(f);
        }
        v16bf bfrag[4];
#pragma unroll
        for (int nt = 0; nt < 4; ++nt) {
            int c = waveN * 64 + nt * 16 + ln16;
            Frag32 f;
            f.lo = *reinterpret_cast<const U128*>(&ldsB[p][c * LDT + hi * 8]);
            f.hi = *reinterpret_cast<const U128*>(&ldsB[p][c * LDT + 16 + hi * 8]);
            bfrag[nt] = frag_cast(f);
        }
#pragma unroll
        for (int mt = 0; mt < 4; ++mt)
#pragma unroll
            for (int nt = 0; nt < 4; ++nt)
                acc[mt][nt] = __builtin_amdgcn_wmma_f32_16x16x32_bf16(
                    false, afrag[mt], false, bfrag[nt],
                    (short)0, acc[mt][nt], false, false);
        // no trailing barrier: next stage writes the other LDS buffer
    }

    // store: C/D element (M = r + 8*hi, N = lane%16); MODE is compile-time.
#pragma unroll
    for (int mt = 0; mt < 4; ++mt)
#pragma unroll
        for (int nt = 0; nt < 4; ++nt)
#pragma unroll
            for (int r = 0; r < 8; ++r) {
                int mg = m0 + waveM * 64 + mt * 16 + r + 8 * hi;
                int ng = n0 + waveN * 64 + nt * 16 + ln16;
                float val = acc[mt][nt][r] + bias[ng];
                if constexpr (MODE == 2) {
                    ((float*)outp)[(size_t)mg * N + ng] = val;
                } else if constexpr (MODE == 0) {
                    ((unsigned short*)outp)[(size_t)mg * N + ng] = f2bf(val);
                } else {  // V transposed: [B,H,DH,S]
                    int b = mg / S, s = mg % S;
                    int h = ng / DH, d = ng % DH;
                    ((unsigned short*)outp)
                        [((size_t)(b * H + h) * DH + d) * S + s] = f2bf(val);
                }
            }
}

// ---------------------------------------------------------------------------
// Flash-style attention with multiplicative mask.
// Grid: (S/256, B*H), 256 threads = 8 waves; wave owns 32 query rows
// (two 16-row tiles sharing every K/V fragment -> 2x arithmetic intensity).
// A block covers 256 q rows of one (b,h): the K/V stream is only re-read
// S/256 = 8 times per head. No block barriers: per-wave LDS transpose
// scratch ordered by s_wait_dscnt (LDS is in-order within a wave).
// Q,K: bf16 [B,S,D] (heads packed), V: bf16 [B,H,DH,S], Out: bf16 [B,S,D].
// ---------------------------------------------------------------------------
__global__ __launch_bounds__(256)
void attention_kernel(const unsigned short* __restrict__ Q,
                      const unsigned short* __restrict__ Kmat,
                      const unsigned short* __restrict__ Vt,
                      const unsigned char*  __restrict__ mask,
                      unsigned short* __restrict__ Out) {
    constexpr int LP = 40;                           // padded stride
    __shared__ __align__(16) unsigned short ldsP[8][2][16 * LP];  // 20 KB

    const int tid  = threadIdx.x;
    const int lane = tid & 31;
    const int wv   = tid >> 5;                       // 0..7
    const int hi   = lane >> 4;
    const int ln16 = lane & 15;
    const int bh = blockIdx.y;
    const int b = bh / H, h = bh % H;
    const int q0 = blockIdx.x * 256 + wv * 32;       // this wave's 32 q rows

    // Q fragments (A layout, row = lane%16) for both q-tiles
    v16bf qfrag[2][2];
#pragma unroll
    for (int qt = 0; qt < 2; ++qt) {
        const unsigned short* qrow =
            Q + ((size_t)(b * S + q0 + qt * 16 + ln16) * D + h * DH);
#pragma unroll
        for (int f = 0; f < 2; ++f) {
            Frag32 t;
            t.lo = *reinterpret_cast<const U128*>(qrow + f * 32 + hi * 8);
            t.hi = *reinterpret_cast<const U128*>(qrow + f * 32 + 16 + hi * 8);
            qfrag[qt][f] = frag_cast(t);
        }
    }

    v8f acc[2][4];
#pragma unroll
    for (int qt = 0; qt < 2; ++qt)
#pragma unroll
        for (int dt = 0; dt < 4; ++dt) acc[qt][dt] = v8f_zero();
    float mrow[2][8], lsum[2][8];
#pragma unroll
    for (int qt = 0; qt < 2; ++qt)
#pragma unroll
        for (int r = 0; r < 8; ++r) { mrow[qt][r] = -1e30f; lsum[qt][r] = 0.f; }

    const unsigned short* kbase_ptr = Kmat + ((size_t)b * S * D + h * DH);
    const unsigned short* vbase_ptr = Vt + ((size_t)(b * H + h) * DH) * S;

    for (int kb = 0; kb < S; kb += 32) {
        // pull the next key/value tiles toward the WGP while we compute
        if (kb + 32 < S) {
            __builtin_prefetch(kbase_ptr + (size_t)(kb + 32 + ln16) * D, 0, 3);
            __builtin_prefetch(vbase_ptr + (size_t)ln16 * S + kb + 32, 0, 3);
        }

        // K^T fragments for this 32-key block, shared by both q-tiles
        v16bf kfrag[2][2];                           // [kt][f]
#pragma unroll
        for (int kt = 0; kt < 2; ++kt) {
            const unsigned short* krow =
                Kmat + ((size_t)(b * S + kb + kt * 16 + ln16) * D + h * DH);
#pragma unroll
            for (int f = 0; f < 2; ++f) {
                Frag32 t;
                t.lo = *reinterpret_cast<const U128*>(krow + f * 32 + hi * 8);
                t.hi = *reinterpret_cast<const U128*>(krow + f * 32 + 16 + hi * 8);
                kfrag[kt][f] = frag_cast(t);
            }
        }

#pragma unroll
        for (int qt = 0; qt < 2; ++qt) {
            float p[2][8];
            float tmax[8];
#pragma unroll
            for (int r = 0; r < 8; ++r) tmax[r] = -1e30f;

#pragma unroll
            for (int kt = 0; kt < 2; ++kt) {
                v8f sc = v8f_zero();
#pragma unroll
                for (int f = 0; f < 2; ++f)
                    sc = __builtin_amdgcn_wmma_f32_16x16x32_bf16(
                        false, qfrag[qt][f], false, kfrag[kt][f],
                        (short)0, sc, false, false);
                // scale 1/sqrt(DH)=0.125 then MULTIPLICATIVE mask
#pragma unroll
                for (int r = 0; r < 8; ++r) {
                    int qg = q0 + qt * 16 + r + 8 * hi;
                    float mv = (float)mask[(size_t)(b * S + qg) * S +
                                           kb + kt * 16 + ln16];
                    float sval = sc[r] * 0.125f * mv;
                    p[kt][r] = sval;
                    tmax[r] = fmaxf(tmax[r], sval);
                }
            }
            // row max across the 16 lanes holding each score row
#pragma unroll
            for (int r = 0; r < 8; ++r) {
                float t = tmax[r];
#pragma unroll
                for (int off = 1; off < 16; off <<= 1)
                    t = fmaxf(t, __shfl_xor(t, off, 16));
                tmax[r] = t;
            }
            // online softmax update
#pragma unroll
            for (int r = 0; r < 8; ++r) {
                float mnew  = fmaxf(mrow[qt][r], tmax[r]);
                float scale = __expf(mrow[qt][r] - mnew);
                mrow[qt][r] = mnew;
                float p0 = __expf(p[0][r] - mnew);
                float p1 = __expf(p[1][r] - mnew);
                p[0][r] = p0; p[1][r] = p1;
                float s = p0 + p1;
#pragma unroll
                for (int off = 1; off < 16; off <<= 1)
                    s += __shfl_xor(s, off, 16);
                lsum[qt][r] = lsum[qt][r] * scale + s;
#pragma unroll
                for (int dt = 0; dt < 4; ++dt) acc[qt][dt][r] *= scale;
            }
            // transpose P (C layout -> A layout) via per-wave LDS scratch
#pragma unroll
            for (int kt = 0; kt < 2; ++kt)
#pragma unroll
                for (int r = 0; r < 8; ++r)
                    ldsP[wv][qt][(r + 8 * hi) * LP + kt * 16 + ln16] =
                        f2bf(p[kt][r]);
        }
        wait_lds();
        v16bf pfrag[2];
#pragma unroll
        for (int qt = 0; qt < 2; ++qt) {
            Frag32 pf;
            pf.lo = *reinterpret_cast<const U128*>(
                &ldsP[wv][qt][ln16 * LP + hi * 8]);
            pf.hi = *reinterpret_cast<const U128*>(
                &ldsP[wv][qt][ln16 * LP + 16 + hi * 8]);
            pfrag[qt] = frag_cast(pf);
        }

        // P @ V : V fragments shared by both q-tiles
#pragma unroll
        for (int dt = 0; dt < 4; ++dt) {
            const unsigned short* vrow =
                Vt + (((size_t)(b * H + h) * DH + dt * 16 + ln16) * S + kb);
            Frag32 t;
            t.lo = *reinterpret_cast<const U128*>(vrow + hi * 8);
            t.hi = *reinterpret_cast<const U128*>(vrow + 16 + hi * 8);
            v16bf vfrag = frag_cast(t);
#pragma unroll
            for (int qt = 0; qt < 2; ++qt)
                acc[qt][dt] = __builtin_amdgcn_wmma_f32_16x16x32_bf16(
                    false, pfrag[qt], false, vfrag,
                    (short)0, acc[qt][dt], false, false);
        }
    }

    // normalize and store merged-head bf16 output [B,S,D]
#pragma unroll
    for (int qt = 0; qt < 2; ++qt)
#pragma unroll
        for (int dt = 0; dt < 4; ++dt)
#pragma unroll
            for (int r = 0; r < 8; ++r) {
                int qg = q0 + qt * 16 + r + 8 * hi;
                float val = acc[qt][dt][r] / lsum[qt][r];
                Out[(size_t)(b * S + qg) * D + h * DH + dt * 16 + ln16] =
                    f2bf(val);
            }
}

// ---------------------------------------------------------------------------
// Launch
// ---------------------------------------------------------------------------
extern "C" void kernel_launch(void* const* d_in, const int* in_sizes, int n_in,
                              void* d_out, int out_size, void* d_ws, size_t ws_size,
                              hipStream_t stream) {
    const float*         query = (const float*)d_in[0];
    const unsigned char* mask  = (const unsigned char*)d_in[1];
    const float* Wq = (const float*)d_in[2];
    const float* bq = (const float*)d_in[3];
    const float* Wk = (const float*)d_in[4];
    const float* bk = (const float*)d_in[5];
    const float* Wv = (const float*)d_in[6];
    const float* bv = (const float*)d_in[7];
    const float* Wo = (const float*)d_in[8];
    const float* bo = (const float*)d_in[9];

    char* ws = (char*)d_ws;
    size_t off = 0;
    auto alloc = [&](size_t bytes) -> void* {
        void* p = ws + off;
        off += (bytes + 255) & ~(size_t)255;
        return p;
    };
    unsigned short* Xb  = (unsigned short*)alloc((size_t)M_TOT * D * 2);
    unsigned short* Wqb = (unsigned short*)alloc((size_t)D * D * 2);
    unsigned short* Wkb = (unsigned short*)alloc((size_t)D * D * 2);
    unsigned short* Wvb = (unsigned short*)alloc((size_t)D * D * 2);
    unsigned short* Wob = (unsigned short*)alloc((size_t)D * D * 2);
    unsigned short* Qb  = (unsigned short*)alloc((size_t)M_TOT * D * 2);
    unsigned short* Kb  = (unsigned short*)alloc((size_t)M_TOT * D * 2);
    unsigned short* Vtb = (unsigned short*)alloc((size_t)M_TOT * D * 2);
    unsigned short* Ab  = (unsigned short*)alloc((size_t)M_TOT * D * 2);

    const int nX = M_TOT * D, nW = D * D;
    f32_to_bf16_kernel<<<(nX / 4 + 255) / 256, 256, 0, stream>>>(query, Xb, nX);
    f32_to_bf16_kernel<<<(nW / 4 + 255) / 256, 256, 0, stream>>>(Wq, Wqb, nW);
    f32_to_bf16_kernel<<<(nW / 4 + 255) / 256, 256, 0, stream>>>(Wk, Wkb, nW);
    f32_to_bf16_kernel<<<(nW / 4 + 255) / 256, 256, 0, stream>>>(Wv, Wvb, nW);
    f32_to_bf16_kernel<<<(nW / 4 + 255) / 256, 256, 0, stream>>>(Wo, Wob, nW);

    dim3 gg(D / 256, M_TOT / 128);
    gemm_bf16_kernel<0><<<gg, 256, 0, stream>>>(Xb, Wqb, bq, Qb,  M_TOT, D, D);
    gemm_bf16_kernel<0><<<gg, 256, 0, stream>>>(Xb, Wkb, bk, Kb,  M_TOT, D, D);
    gemm_bf16_kernel<1><<<gg, 256, 0, stream>>>(Xb, Wvb, bv, Vtb, M_TOT, D, D);

    attention_kernel<<<dim3(S / 256, Bb * H), 256, 0, stream>>>(Qb, Kb, Vtb, mask, Ab);

    gemm_bf16_kernel<2><<<gg, 256, 0, stream>>>(Ab, Wob, bo, d_out, M_TOT, D, D);
}
